// RAE_73718818668904
// MI455X (gfx1250) — compile-verified, matrix operand order checked by reading, not run on hardware
//
#include <hip/hip_runtime.h>
#include <hip/hip_bf16.h>

typedef _Float16 h16 __attribute__((ext_vector_type(16)));
typedef _Float16 h8  __attribute__((ext_vector_type(8)));
typedef float    f8  __attribute__((ext_vector_type(8)));

#define EMB   128
#define BATCH 16
#define SEQ   128
#define EPSV  1e-5f

// packed f16 weight offsets (elements) inside d_ws
#define OFF_W1 0
#define OFF_W2 (OFF_W1 + 256*192)
#define OFF_W3 (OFF_W2 + 192*128)
#define OFF_W4 (OFF_W3 + 128*192)

// ---- LDS layout (bytes) ----------------------------------------------------
#define LDS_W1     0          // 98304  B  (256x192 f16, fragment-packed)
#define LDS_W2     98304      // 49152  B  (192x128 f16)
#define LDS_W3     147456     // 49152  B  (128x192 f16)
#define WSTAGE_BYTES 196608   // W1+W2+W3 staged in one TDM transfer
#define LDS_NODES  196608     // 32768  B  (128x128 f16)
#define LDS_Y      229376     // 32768  B  (128x128 f16)
#define LDS_SCR    262144     // 49152  B  (4 waves x 12288)
#define SCR_PER_WAVE 12288    // h1/h3: 16x192 f32; xx: 16x256 f16 (overlaps)
#define LDS_LOSS   311296     // 512 B
#define LDS_CNT    311808     // 512 B
#define LDS_IDX    312320
#define LDS_TOTAL  312448

// ---------------------------------------------------------------------------
// Pack f32 weight [K][N] (row-major) into WMMA B-fragment-major f16:
//   P[(((kt*NT + nt)*32 + lane)*16 + e)] = W[k][n]
//   n = nt*16 + (lane&15);  k = kt*32 + ((lane&16)?8:0) + (e&7) + ((e&8)<<1)
// ---------------------------------------------------------------------------
__global__ void rae_pack_w(const float* __restrict__ W, _Float16* __restrict__ P,
                           int K, int N) {
  int i = blockIdx.x * blockDim.x + threadIdx.x;
  if (i >= K * N) return;
  int e    = i & 15;
  int lane = (i >> 4) & 31;
  int rem  = i >> 9;
  int NT   = N >> 4;
  int nt   = rem % NT;
  int kt   = rem / NT;
  int k = kt * 32 + ((lane & 16) ? 8 : 0) + (e & 7) + ((e & 8) << 1);
  int n = nt * 16 + (lane & 15);
  P[i] = (_Float16)W[k * N + n];
}

__global__ void rae_init_out(float* out) {
  if (threadIdx.x == 0 && blockIdx.x == 0) out[BATCH] = 0.0f;
}

// ---------------- fragment helpers (CDNA5 wave32 WMMA layouts) -------------
__device__ __forceinline__ f8 wmma_f16(h16 a, h16 b, f8 c) {
  return __builtin_amdgcn_wmma_f32_16x16x32_f16(false, a, false, b,
                                                (short)0, c, false, false);
}

// works for LDS (ds_load_b128) or global (global_load_b128) packed weights
__device__ __forceinline__ h16 load_b_pack(const _Float16* P, int NT,
                                           int kt, int nt, int lane) {
  const _Float16* p = P + ((((kt * NT + nt) << 5) + lane) << 4);
  h8 lo = *(const h8*)(p);
  h8 hi = *(const h8*)(p + 8);
  h16 f;
#pragma unroll
  for (int e = 0; e < 8; ++e) { f[e] = lo[e]; f[e + 8] = hi[e]; }
  return f;
}

// A fragment from concat([nodes[j+1], nodes[j]]): kt<4 -> right, kt>=4 -> left
__device__ __forceinline__ h16 load_a_concat(const _Float16* sN,
                                             int j0, int kt, int lane) {
  int row = j0 + (lane & 15);
  if (row > 126) row = 126;
  const _Float16* src = (kt < 4) ? (sN + (row + 1) * EMB) : (sN + row * EMB);
  int kb = ((kt & 3) << 5) + ((lane & 16) ? 8 : 0);
  h8 lo = *(const h8*)(src + kb);
  h8 hi = *(const h8*)(src + kb + 16);
  h16 f;
#pragma unroll
  for (int e = 0; e < 8; ++e) { f[e] = lo[e]; f[e + 8] = hi[e]; }
  return f;
}

__device__ __forceinline__ h16 load_a_h16(const _Float16* base, int ld,
                                          int kt, int lane) {
  const _Float16* src = base + (lane & 15) * ld + (kt << 5) + ((lane & 16) ? 8 : 0);
  h8 lo = *(const h8*)(src);
  h8 hi = *(const h8*)(src + 16);
  h16 f;
#pragma unroll
  for (int e = 0; e < 8; ++e) { f[e] = lo[e]; f[e + 8] = hi[e]; }
  return f;
}

// A fragment from f32 wave scratch (leading dim ld), f32 -> f16 on the fly
__device__ __forceinline__ h16 load_a_f32(const float* base, int ld,
                                          int kt, int lane) {
  const float* src = base + (lane & 15) * ld + (kt << 5) + ((lane & 16) ? 8 : 0);
  h16 f;
#pragma unroll
  for (int e = 0; e < 8; ++e) {
    f[e]     = (_Float16)src[e];
    f[e + 8] = (_Float16)src[e + 16];
  }
  return f;
}

// ---------------------------------------------------------------------------
// Persistent scan: one block per batch, 128 threads (4 wave32).
// W1/W2/W3 staged to LDS once via TDM; W4 streamed from L2-resident pack.
// ---------------------------------------------------------------------------
__global__ void __launch_bounds__(128)
rae_scan(const int* __restrict__ inp, const float* __restrict__ emb,
         const float* __restrict__ b1, const float* __restrict__ b2,
         const float* __restrict__ b3, const float* __restrict__ b4,
         const float* __restrict__ Wk, const float* __restrict__ bk,
         const _Float16* __restrict__ pack, float* __restrict__ out) {
  extern __shared__ char smem[];
  const _Float16* sW1 = (const _Float16*)(smem + LDS_W1);
  const _Float16* sW2 = (const _Float16*)(smem + LDS_W2);
  const _Float16* sW3 = (const _Float16*)(smem + LDS_W3);
  _Float16* sN    = (_Float16*)(smem + LDS_NODES);
  _Float16* sY    = (_Float16*)(smem + LDS_Y);
  float*    sLoss = (float*)(smem + LDS_LOSS);
  float*    sCnt  = (float*)(smem + LDS_CNT);
  int*      sIdx  = (int*)(smem + LDS_IDX);

  const int b    = blockIdx.x;
  const int tid  = threadIdx.x;          // 0..127 == node column
  const int lane = tid & 31;
  const int wave = tid >> 5;             // 0..3
  float*    myH  = (float*)(smem + LDS_SCR + wave * SCR_PER_WAVE);    // 16x192 f32
  _Float16* myXX = (_Float16*)(smem + LDS_SCR + wave * SCR_PER_WAVE); // 16x256 f16

  const _Float16* P4 = pack + OFF_W4;
  if (tid == 0) __builtin_prefetch(P4, 0, 3);   // warm streamed-W4 path

  // ---- stage W1..W3 (196608 B) into LDS with the Tensor Data Mover -------
#if __has_builtin(__builtin_amdgcn_tensor_load_to_lds)
  if (wave == 0) {
    typedef unsigned int u32x4 __attribute__((ext_vector_type(4)));
    typedef int          i32x4 __attribute__((ext_vector_type(4)));
    typedef int          i32x8 __attribute__((ext_vector_type(8)));
    unsigned long long ga = (unsigned long long)(uintptr_t)pack;
    unsigned int lds = (unsigned int)(uintptr_t)(smem + LDS_W1); // LDS byte offset
    const int NDW = WSTAGE_BYTES / 4;          // 49152 dwords, one 1-row tile
    u32x4 g0;
    g0[0] = 1u;                                 // count=1 (valid), no gather
    g0[1] = lds;                                // lds_addr
    g0[2] = (unsigned int)ga;                   // global_addr[31:0]
    g0[3] = (unsigned int)((ga >> 32) & 0x01FFFFFFull) | 0x80000000u; // [56:32] | type=2
    i32x8 g1;
    g1[0] = (2 << 16);                          // data_size = 4 bytes
    g1[1] = (int)((NDW & 0xFFFF) << 16);        // tensor_dim0[15:0]
    g1[2] = (NDW >> 16) | (1 << 16);            // tensor_dim0[31:16] | tensor_dim1=1
    g1[3] = (int)((unsigned)NDW << 16);         // tile_dim0
    g1[4] = 1;                                  // tile_dim1=1, tile_dim2=0
    g1[5] = NDW;                                // tensor_dim0_stride
    g1[6] = 0;
    g1[7] = 0;
    i32x4 gz = {0, 0, 0, 0};
#if __clang_major__ >= 23
    i32x8 gz8 = {0, 0, 0, 0, 0, 0, 0, 0};
    __builtin_amdgcn_tensor_load_to_lds(g0, g1, gz, gz, gz8, 0);
#else
    __builtin_amdgcn_tensor_load_to_lds(g0, g1, gz, gz, 0);
#endif
    __builtin_amdgcn_s_wait_tensorcnt((short)0);
  }
#else
  for (int i = tid; i < WSTAGE_BYTES / 4; i += 128)
    ((unsigned int*)smem)[i] = ((const unsigned int*)pack)[i];
#endif

  // ---- embedding gather: nodes[s][tid] = emb[inputs[b][s]][tid] (f16) ----
  for (int s = 0; s < SEQ; ++s) {
    int tok = inp[b * SEQ + s];
    sN[s * EMB + tid] = (_Float16)emb[tok * EMB + tid];
  }
  sCnt[tid] = 1.0f;
  float accT = 0.0f;
  __syncthreads();   // weights staged + state initialized

  for (int t = 0; t < SEQ - 1; ++t) {
    const int L = SEQ - t;
    const int nvalid = L - 1;
    const int ntiles = (nvalid + 15) >> 4;

    // -------- phase 1: GEMM chain + loss per 16-row tile ------------------
    for (int tile = wave; tile < ntiles; tile += 4) {
      const int j0 = tile << 4;

      // GEMM1: concat(right,left)(16x256) @ W1(256x192) + b1 -> h1 (f32)
      h16 a1[8];
#pragma unroll
      for (int kt = 0; kt < 8; ++kt) a1[kt] = load_a_concat(sN, j0, kt, lane);
      for (int nt = 0; nt < 12; ++nt) {
        f8 c = {};
#pragma unroll
        for (int kt = 0; kt < 8; ++kt)
          c = wmma_f16(a1[kt], load_b_pack(sW1, 12, kt, nt, lane), c);
        int n = (nt << 4) + (lane & 15);
        float bias = b1[n];
        int mb = (lane & 16) ? 8 : 0;
#pragma unroll
        for (int r = 0; r < 8; ++r) myH[(r + mb) * 192 + n] = c[r] + bias;
      }

      // GEMM2: h1(16x192) @ W2(192x128) + b2 -> y (f16 in sY)
      h16 a2[6];
#pragma unroll
      for (int kt = 0; kt < 6; ++kt) a2[kt] = load_a_f32(myH, 192, kt, lane);
      for (int nt = 0; nt < 8; ++nt) {
        f8 c = {};
#pragma unroll
        for (int kt = 0; kt < 6; ++kt)
          c = wmma_f16(a2[kt], load_b_pack(sW2, 8, kt, nt, lane), c);
        int n = (nt << 4) + (lane & 15);
        float bias = b2[n];
        int mb = (lane & 16) ? 8 : 0;
#pragma unroll
        for (int r = 0; r < 8; ++r)
          sY[(j0 + r + mb) * EMB + n] = (_Float16)(c[r] + bias);
      }

      // GEMM3: y(16x128) @ W3(128x192) + b3 -> h3 (f32, reuses h1)
      h16 a3[4];
#pragma unroll
      for (int kt = 0; kt < 4; ++kt) a3[kt] = load_a_h16(sY + j0 * EMB, EMB, kt, lane);
      for (int nt = 0; nt < 12; ++nt) {
        f8 c = {};
#pragma unroll
        for (int kt = 0; kt < 4; ++kt)
          c = wmma_f16(a3[kt], load_b_pack(sW3, 12, kt, nt, lane), c);
        int n = (nt << 4) + (lane & 15);
        float bias = b3[n];
        int mb = (lane & 16) ? 8 : 0;
#pragma unroll
        for (int r = 0; r < 8; ++r) myH[(r + mb) * 192 + n] = c[r] + bias;
      }

      // GEMM4: h3(16x192) @ W4(192x256) + b4 -> xx (f16, overlaps h3)
      h16 a4[6];
#pragma unroll
      for (int kt = 0; kt < 6; ++kt) a4[kt] = load_a_f32(myH, 192, kt, lane);
      for (int nt = 0; nt < 16; ++nt) {
        f8 c = {};
#pragma unroll
        for (int kt = 0; kt < 6; ++kt)
          c = wmma_f16(a4[kt], load_b_pack(P4, 16, kt, nt, lane), c);
        int n = (nt << 4) + (lane & 15);
        float bias = b4[n];
        int mb = (lane & 16) ? 8 : 0;
#pragma unroll
        for (int r = 0; r < 8; ++r)
          myXX[(r + mb) * 256 + n] = (_Float16)(c[r] + bias);
      }

      // log-softmax + weighted recon losses:
      //   lanes 0..15 -> ls1 (cols 0..127 vs nodes[j+1]); 16..31 -> ls2
      {
        int half = lane >> 4;
        int m    = lane & 15;
        int j    = j0 + m;
        int jr   = (j > 126) ? 126 : j;
        const _Float16* xrow = myXX + m * 256 + half * 128;
        float mx = -__builtin_inff();
        for (int k = 0; k < 128; ++k) mx = fmaxf(mx, (float)xrow[k]);
        float se = 0.0f;
        for (int k = 0; k < 128; ++k) se += __expf((float)xrow[k] - mx);
        float lse = __logf(se) + mx;
        const _Float16* nrow = sN + ((half == 0) ? (jr + 1) : jr) * EMB;
        float s = 0.0f;
        for (int k = 0; k < 128; ++k) s += (float)nrow[k] * ((float)xrow[k] - lse);
        float ls = -s;
        float other = __shfl_xor(ls, 16, 32);   // pair ls1 <-> ls2
        if (half == 0 && j < nvalid) {
          float n1 = sCnt[j + 1], n2 = sCnt[j];
          sLoss[j] = ((n1 + EPSV) * ls + (n2 + EPSV) * other) / (n1 + n2 + EPSV);
        }
      }
    }
    __syncthreads();

    // -------- phase 2: argmin + counts update (thread 0, serial) ----------
    if (tid == 0) {
      float best = __builtin_inff();
      int bi = 0;
      for (int j = 0; j < nvalid; ++j) {
        float v = sLoss[j];
        if (v < best) { best = v; bi = j; }      // first-min tie semantics
      }
      accT += best;
      int   lc = (bi == 0) ? (L - 1) : (bi - 1);
      float fc = sCnt[bi] + sCnt[lc];            // father count (pre-shift)
      int  pos = (bi > 0) ? (bi - 1) : 0;
      if (bi > 0)
        for (int k = bi - 1; k <= L - 2; ++k) sCnt[k] = sCnt[k + 1];
      sCnt[pos] = fc;
      *sIdx = bi;
    }
    __syncthreads();

    // -------- phase 3: node left-shift + father insert (col-per-thread) ---
    {
      int bi  = *sIdx;
      int pos = (bi > 0) ? (bi - 1) : 0;
      if (bi > 0) {
        for (int k = bi - 1; k <= L - 2; ++k)
          sN[k * EMB + tid] = sN[(k + 1) * EMB + tid];
      }
      sN[pos * EMB + tid] = sY[bi * EMB + tid];  // father_x = y[idx]
    }
    __syncthreads();
  }

  // ---- classifier head + loss reduction ----------------------------------
  if (tid == 0) {
    float best = -__builtin_inff();
    int bc = 0;
    for (int c = 0; c < 5; ++c) {
      float s = bk[c];
      for (int d = 0; d < EMB; ++d) s += (float)sN[d] * Wk[d * 5 + c];
      if (s > best) { best = s; bc = c; }        // first-max tie semantics
    }
    out[b] = (float)bc;
    atomicAdd(out + BATCH, accT / (float)BATCH);
  }
}

// ---------------------------------------------------------------------------
extern "C" void kernel_launch(void* const* d_in, const int* in_sizes, int n_in,
                              void* d_out, int out_size, void* d_ws, size_t ws_size,
                              hipStream_t stream) {
  (void)in_sizes; (void)n_in; (void)out_size; (void)ws_size;
  const int*   inp = (const int*)d_in[0];
  const float* emb = (const float*)d_in[1];
  const float* W1  = (const float*)d_in[2];
  const float* b1  = (const float*)d_in[3];
  const float* W2  = (const float*)d_in[4];
  const float* b2  = (const float*)d_in[5];
  const float* W3  = (const float*)d_in[6];
  const float* b3  = (const float*)d_in[7];
  const float* W4  = (const float*)d_in[8];
  const float* b4  = (const float*)d_in[9];
  const float* Wk  = (const float*)d_in[10];
  const float* bk  = (const float*)d_in[11];
  float*    out  = (float*)d_out;
  _Float16* pack = (_Float16*)d_ws;

  rae_init_out<<<1, 32, 0, stream>>>(out);
  rae_pack_w<<<(256 * 192 + 255) / 256, 256, 0, stream>>>(W1, pack + OFF_W1, 256, 192);
  rae_pack_w<<<(192 * 128 + 255) / 256, 256, 0, stream>>>(W2, pack + OFF_W2, 192, 128);
  rae_pack_w<<<(128 * 192 + 255) / 256, 256, 0, stream>>>(W3, pack + OFF_W3, 128, 192);
  rae_pack_w<<<(192 * 256 + 255) / 256, 256, 0, stream>>>(W4, pack + OFF_W4, 192, 256);

  rae_scan<<<BATCH, 128, LDS_TOTAL, stream>>>(inp, emb, b1, b2, b3, b4,
                                              Wk, bk, pack, out);
}
